// SCNUTrain_62912680951980
// MI455X (gfx1250) — matrix operand order repaired and checked by reading, not compiled, not run on hardware
//
#include <hip/hip_runtime.h>
#include <cstdint>
#include <cstddef>

// ---------------------------------------------------------------------------
// SCN (Stochastic Configuration Network) greedy training on MI455X (gfx1250).
//
//  Step-dominant work: H = sigmoid(X[10000x784] @ Wcand[784x150] + b), 64x.
//  -> bf16 V_WMMA_F32_16X16X32_BF16 with f32 accumulate; X cached in L2,
//     W panel resident in LDS (256KB of the WGP's 320KB), X tiles streamed
//     into LDS by the Tensor Data Mover (double buffered, s_wait_tensorcnt).
//  Everything else (Gram-Schmidt QR refit, triangular solve, residual) is
//  tiny fp32 VALU work in small helper kernels, sequenced on the stream.
// ---------------------------------------------------------------------------

#define NROW   10000
#define KDIM   784
#define KP     800          // K padded to 25 * 32
#define KSTEPS 25
#define CAND   160          // 150 candidates padded to 10 N-tiles
#define CREAL  150
#define MAXN   64
#define MCLS   10
#define MT     5            // M-tiles (16 rows) per block
#define NBLK   125          // 125 * 5 * 16 = 10000 rows
#define GEMM_THREADS 320    // 10 wave32s, one 16-col N-tile each

typedef float        v8f   __attribute__((ext_vector_type(8)));
typedef __bf16       v16bf __attribute__((ext_vector_type(16)));
typedef unsigned int u32v4 __attribute__((ext_vector_type(4)));
typedef float        f32v4 __attribute__((ext_vector_type(4)));
typedef unsigned int tdm_u32x4 __attribute__((ext_vector_type(4)));
typedef int          tdm_i32x4 __attribute__((ext_vector_type(4)));
typedef int          tdm_i32x8 __attribute__((ext_vector_type(8)));

union FragBF { u32v4 u[2]; v16bf v; };   // 16 bf16 = 8 dwords (A or B fragment)

// LDS layout for the GEMM kernel
#define SM_W_BYTES (CAND * KP * 2)            // 256000
#define SM_X_BYTES (16 * KP * 2)              // 25600 per buffer
#define SM_E_OFF   (SM_W_BYTES + 2 * SM_X_BYTES)
#define SMEM_BYTES (SM_E_OFF + 16 * MCLS * 4) // 307200 + 640 <= 320KB

// Use TDM only in the device pass and only if the builtin exists.
#if defined(__HIP_DEVICE_COMPILE__) && defined(__has_builtin)
# if __has_builtin(__builtin_amdgcn_tensor_load_to_lds) && __has_builtin(__builtin_amdgcn_s_wait_tensorcnt)
#  define USE_TDM 1
# endif
#endif
#ifndef USE_TDM
# define USE_TDM 0
#endif

#if USE_TDM
// Build a 2D D# (data_size = 2 bytes) and fire TENSOR_LOAD_TO_LDS.
__device__ __forceinline__ void tdm_load_2d(unsigned lds_byte_off, const void* gptr,
                                            unsigned td0, unsigned td1,
                                            unsigned tl0, unsigned tl1,
                                            unsigned long long stride0)
{
  unsigned long long ga = (unsigned long long)(uintptr_t)gptr;
  tdm_u32x4 g0;
  g0[0] = 1u;                                            // count=1, is_restore=0
  g0[1] = lds_byte_off;                                  // D#.lds_addr
  g0[2] = (unsigned)(ga & 0xFFFFFFFFull);                // global_addr[31:0]
  g0[3] = (unsigned)((ga >> 32) & 0x1FFFFFFull) | (2u << 30); // addr[56:32] | type=2
  tdm_i32x8 g1;
  g1[0] = (int)(1u << 16);                               // data_size = 2 bytes
  g1[1] = (int)((td0 & 0xFFFFu) << 16);                  // tensor_dim0[15:0]
  g1[2] = (int)(((td0 >> 16) & 0xFFFFu) | ((td1 & 0xFFFFu) << 16));
  g1[3] = (int)(((td1 >> 16) & 0xFFFFu) | ((tl0 & 0xFFFFu) << 16)); // tile_dim0
  g1[4] = (int)(tl1 & 0xFFFFu);                          // tile_dim1 (tile_dim2=0)
  g1[5] = (int)(stride0 & 0xFFFFFFFFull);                // tensor_dim0_stride
  g1[6] = (int)((stride0 >> 32) & 0xFFFFull);
  g1[7] = 0;
  tdm_i32x4 z4 = {0, 0, 0, 0};
#if defined(__clang_major__) && (__clang_major__ >= 23)
  tdm_i32x8 z8 = {0, 0, 0, 0, 0, 0, 0, 0};
  __builtin_amdgcn_tensor_load_to_lds(g0, g1, z4, z4, z8, 0);
#else
  __builtin_amdgcn_tensor_load_to_lds(g0, g1, z4, z4, 0);
#endif
}
#endif

__device__ __forceinline__ float warp_sum32(float v) {
#pragma unroll
  for (int off = 16; off > 0; off >>= 1) v += __shfl_xor(v, off, 32);
  return v;
}

__device__ __forceinline__ unsigned pcg_hash(unsigned v) {
  v = v * 747796405u + 2891336453u;
  unsigned w = ((v >> ((v >> 28) + 4u)) ^ v) * 277803737u;
  return (w >> 22) ^ w;
}

// --------------------------- helper kernels --------------------------------

__global__ void scn_cvt_x(const float* __restrict__ X, __bf16* __restrict__ Xb) {
  int idx = blockIdx.x * blockDim.x + threadIdx.x;
  if (idx >= NROW * KP) return;
  int r = idx / KP, d = idx % KP;
  Xb[idx] = (__bf16)((d < KDIM) ? X[(size_t)r * KDIM + d] : 0.0f);
}

__global__ void scn_copy(float* __restrict__ dst, const float* __restrict__ src, int n) {
  int i = blockIdx.x * blockDim.x + threadIdx.x;
  if (i < n) dst[i] = src[i];
}

__global__ void scn_zero(float* __restrict__ p, int n) {
  int i = blockIdx.x * blockDim.x + threadIdx.x;
  if (i < n) p[i] = 0.0f;
}

// Candidate generation: Wt[160][800] bf16 (transposed, K-contiguous) + b[160].
// scale = {0.5,1.0,5.0} per 50-candidate lambda block; padded rows/cols = 0.
__global__ void scn_gen(__bf16* __restrict__ Wtb, float* __restrict__ bvec, int step) {
  int idx = blockIdx.x * blockDim.x + threadIdx.x;
  if (idx >= CAND * KP) return;
  int c = idx / KP, d = idx % KP;
  float scale = (c < 50) ? 0.5f : (c < 100) ? 1.0f : (c < CREAL) ? 5.0f : 0.0f;
  unsigned h = pcg_hash((unsigned)idx * 2654435761u ^ (unsigned)(step * 0x9E3779B9) ^ 0xA511E9B3u);
  float u = (float)(h >> 8) * (1.0f / 16777216.0f);
  float val = (d < KDIM) ? scale * (2.0f * u - 1.0f) : 0.0f;
  Wtb[idx] = (__bf16)val;
  if (d == 0) {
    unsigned hb = pcg_hash((unsigned)c * 0x85EBCA6Bu ^ (unsigned)(step * 0xC2B2AE35) ^ 0x27D4EB2Fu);
    float ub = (float)(hb >> 8) * (1.0f / 16777216.0f);
    bvec[c] = scale * (2.0f * ub - 1.0f);
  }
}

// ------------------------- main WMMA GEMM kernel ---------------------------
// Hc[10000][160] = sigmoid(Xb[10000][800] @ Wt^T + b); also accumulates
// ssq[c] = sum_i h^2 and num[j][c] = sum_i e[i][j] h[i][c] via atomics.
__global__ void __launch_bounds__(GEMM_THREADS, 1)
scn_gemm(const __bf16* __restrict__ Xb, const __bf16* __restrict__ Wtb,
         const float* __restrict__ bvec, const float* __restrict__ Ein,
         float* __restrict__ Hc, float* __restrict__ ssq, float* __restrict__ num)
{
  extern __shared__ __align__(16) unsigned char smem[];
  __bf16* sW  = (__bf16*)smem;                               // [160][800] bf16
  __bf16* sX0 = (__bf16*)(smem + SM_W_BYTES);                // [16][800] bf16
  __bf16* sX1 = (__bf16*)(smem + SM_W_BYTES + SM_X_BYTES);
  float*  sE  = (float*)(smem + SM_E_OFF);                   // [10][16] f32 (transposed tile)

  const int tid  = threadIdx.x;
  const int lane = tid & 31;
  const int wv   = tid >> 5;        // N-tile owner
  const int g    = lane >> 4;       // lane half (K/M split per ISA layout)
  const int l16  = lane & 15;
  const int col  = wv * 16 + l16;   // candidate column
  const int mt0  = blockIdx.x * MT;

  const float bias = bvec[col];

#if USE_TDM
  if (wv == 0) {
    // whole W panel -> LDS (256KB), then first X tile; both tracked by TENSORcnt
    tdm_load_2d((unsigned)(size_t)sW, Wtb, KP, CAND, KP, CAND, KP);
    tdm_load_2d((unsigned)(size_t)sX0, Xb + (size_t)mt0 * 16 * KP, KP, 16, KP, 16, KP);
  }
#else
  {
    const u32v4* src = (const u32v4*)Wtb;
    u32v4* dst = (u32v4*)sW;
    for (int q = tid; q < SM_W_BYTES / 16; q += GEMM_THREADS) dst[q] = src[q];
  }
#endif

  float lssq = 0.0f;
  float lnum[MCLS];
#pragma unroll
  for (int j = 0; j < MCLS; ++j) lnum[j] = 0.0f;

  for (int t = 0; t < MT; ++t) {
    const int mt = mt0 + t;
    __bf16* xs = (t & 1) ? sX1 : sX0;

#if USE_TDM
    if (wv == 0) {
      if (t + 1 < MT) {                       // prefetch next X tile, keep 1 in flight
        __bf16* nb = ((t + 1) & 1) ? sX1 : sX0;
        tdm_load_2d((unsigned)(size_t)nb, Xb + (size_t)(mt + 1) * 16 * KP, KP, 16, KP, 16, KP);
        __builtin_amdgcn_s_wait_tensorcnt(1);
      } else {
        __builtin_amdgcn_s_wait_tensorcnt(0);
      }
    }
#else
    {
      const u32v4* src = (const u32v4*)(Xb + (size_t)mt * 16 * KP);
      u32v4* dst = (u32v4*)xs;
      for (int q = tid; q < SM_X_BYTES / 16; q += GEMM_THREADS) dst[q] = src[q];
    }
#endif
    if (tid < 16 * MCLS) {                    // residual tile, transposed [j][row]
      int rr = tid / MCLS, jj = tid % MCLS;
      sE[jj * 16 + rr] = Ein[(size_t)(mt * 16 + rr) * MCLS + jj];
    }
    __syncthreads();                          // LDS (TDM + sE) visible to all waves

    v8f acc;
#pragma unroll
    for (int r = 0; r < 8; ++r) acc[r] = bias;   // bias depends on N only

    // A: row l16 of X tile; halves at k+{0..7} / k+{16..23} (+8 for lanes 16-31)
    const __bf16* xrow = xs + l16 * KP + g * 8;
    // B: candidate row `col`; halves at k+{0..7} / k+{16..23} (+8 for lanes 16-31)
    const __bf16* wrow = sW + col * KP + g * 8;
#pragma unroll
    for (int ks = 0; ks < KSTEPS; ++ks) {
      FragBF a, b;
      a.u[0] = *(const u32v4*)(xrow + ks * 32);
      a.u[1] = *(const u32v4*)(xrow + ks * 32 + 16);
      b.u[0] = *(const u32v4*)(wrow + ks * 32);
      b.u[1] = *(const u32v4*)(wrow + ks * 32 + 16);
      acc = __builtin_amdgcn_wmma_f32_16x16x32_bf16(false, a.v, false, b.v,
                                                    (short)0, acc, false, false);
    }

    float h[8];
#pragma unroll
    for (int r = 0; r < 8; ++r) h[r] = 1.0f / (1.0f + __expf(-acc[r]));

    const int rowbase = mt * 16 + g * 8;      // C/D layout: VGPR r -> M = r + 8*g
#pragma unroll
    for (int r = 0; r < 8; ++r) {
      Hc[(size_t)(rowbase + r) * CAND + col] = h[r];
      lssq += h[r] * h[r];
    }
#pragma unroll
    for (int j = 0; j < MCLS; ++j) {
      const f32v4* ep = (const f32v4*)(sE + j * 16 + g * 8);
      f32v4 e0 = ep[0], e1 = ep[1];
      lnum[j] += h[0]*e0.x + h[1]*e0.y + h[2]*e0.z + h[3]*e0.w
               + h[4]*e1.x + h[5]*e1.y + h[6]*e1.z + h[7]*e1.w;
    }
    __syncthreads();                          // safe to overwrite staged LDS
  }

  // fold the lane/lane+16 halves of each column, then one atomic per column
  lssq += __shfl_xor(lssq, 16, 32);
  if (lane < 16) atomicAdd(&ssq[col], lssq);
#pragma unroll
  for (int j = 0; j < MCLS; ++j) {
    float s = lnum[j] + __shfl_xor(lnum[j], 16, 32);
    if (lane < 16) atomicAdd(&num[j * CAND + col], s);
  }
}

// --------------------- selection + QR refit kernels ------------------------

__global__ void scn_select(const float* __restrict__ num, const float* __restrict__ ssq,
                           const __bf16* __restrict__ Wtb, const float* __restrict__ bvec,
                           float* __restrict__ Wout, float* __restrict__ bout,
                           int* __restrict__ best, int step)
{
  __shared__ float sv[256];
  __shared__ int   si[256];
  __shared__ int   sbest;
  int tid = threadIdx.x;
  float v = -1.0f;
  if (tid < CREAL) {
    float s = ssq[tid], a = 0.0f;
#pragma unroll
    for (int j = 0; j < MCLS; ++j) { float n = num[j * CAND + tid]; a += n * n; }
    v = a / (10.0f * s);
  }
  sv[tid] = v; si[tid] = tid;
  __syncthreads();
  for (int off = 128; off > 0; off >>= 1) {
    if (tid < off && sv[tid + off] > sv[tid]) { sv[tid] = sv[tid + off]; si[tid] = si[tid + off]; }
    __syncthreads();
  }
  if (tid == 0) { sbest = si[0]; best[0] = sbest; bout[step] = bvec[sbest]; }
  __syncthreads();
  int bc = sbest;
  for (int d = tid; d < KDIM; d += 256)
    Wout[(size_t)d * MAXN + step] = (float)Wtb[(size_t)bc * KP + d];
}

__global__ void scn_gather(float* __restrict__ H, const float* __restrict__ Hc,
                           const int* __restrict__ best, int step)
{
  int stride = gridDim.x * blockDim.x;
  int bc = best[0];
  for (int i = blockIdx.x * blockDim.x + threadIdx.x; i < NROW; i += stride)
    H[(size_t)step * NROW + i] = Hc[(size_t)i * CAND + bc];
}

// r1 = Q[0..k-1]^T h_k
__global__ void scn_proj(float* __restrict__ r1, const float* __restrict__ Q,
                         const float* __restrict__ H, int k)
{
  const float* hk = H + (size_t)k * NROW;
  int stride = gridDim.x * blockDim.x;
  int i0 = blockIdx.x * blockDim.x + threadIdx.x;
  for (int j = 0; j < k; ++j) {
    const float* qj = Q + (size_t)j * NROW;
    float s = 0.0f;
    for (int i = i0; i < NROW; i += stride) s += qj[i] * hk[i];
    s = warp_sum32(s);
    if ((threadIdx.x & 31) == 0) atomicAdd(&r1[j], s);
  }
}

// t = h_k - Q r1 (stored unnormalized into Q[k]); scal[0] += ||t||^2
__global__ void scn_maket(float* __restrict__ Q, const float* __restrict__ H,
                          const float* __restrict__ r1, float* __restrict__ scal, int k)
{
  const float* hk = H + (size_t)k * NROW;
  float* qk = Q + (size_t)k * NROW;
  int stride = gridDim.x * blockDim.x;
  float ts = 0.0f;
  for (int i = blockIdx.x * blockDim.x + threadIdx.x; i < NROW; i += stride) {
    float tv = hk[i];
    for (int j = 0; j < k; ++j) tv -= Q[(size_t)j * NROW + i] * r1[j];
    qk[i] = tv;
    ts += tv * tv;
  }
  ts = warp_sum32(ts);
  if ((threadIdx.x & 31) == 0) atomicAdd(&scal[0], ts);
}

// append column to R, compute 1/r2, zero QT row k
__global__ void scn_finr(float* __restrict__ R, float* __restrict__ QT,
                         const float* __restrict__ r1, float* __restrict__ scal, int k)
{
  int tid = threadIdx.x;
  if (tid == 0) {
    float r2 = sqrtf(scal[0]);
    R[(size_t)k * MAXN + k] = r2;
    scal[1] = (r2 > 0.0f) ? 1.0f / r2 : 0.0f;
  }
  if (tid < k)    R[(size_t)tid * MAXN + k] = r1[tid];
  if (tid < MCLS) QT[(size_t)k * MCLS + tid] = 0.0f;
}

// q = t / r2; QT[k] = q^T Y
__global__ void scn_normq(float* __restrict__ Q, const float* __restrict__ Y,
                          float* __restrict__ QT, const float* __restrict__ scal, int k)
{
  const float inv = scal[1];
  float* qk = Q + (size_t)k * NROW;
  float accj[MCLS];
#pragma unroll
  for (int j = 0; j < MCLS; ++j) accj[j] = 0.0f;
  int stride = gridDim.x * blockDim.x;
  for (int i = blockIdx.x * blockDim.x + threadIdx.x; i < NROW; i += stride) {
    float q = qk[i] * inv;
    qk[i] = q;
#pragma unroll
    for (int j = 0; j < MCLS; ++j) accj[j] += q * Y[(size_t)i * MCLS + j];
  }
#pragma unroll
  for (int j = 0; j < MCLS; ++j) {
    float s = warp_sum32(accj[j]);
    if ((threadIdx.x & 31) == 0) atomicAdd(&QT[(size_t)k * MCLS + j], s);
  }
}

// back-substitution: R beta = QT, one thread per RHS column
__global__ void scn_solve(const float* __restrict__ R, const float* __restrict__ QT,
                          float* __restrict__ beta, int k)
{
  int m = threadIdx.x;
  if (m >= MCLS) return;
  for (int row = k; row >= 0; --row) {
    float s = QT[(size_t)row * MCLS + m];
    for (int t = row + 1; t <= k; ++t) s -= R[(size_t)row * MAXN + t] * beta[(size_t)t * MCLS + m];
    float d = R[(size_t)row * MAXN + row];
    beta[(size_t)row * MCLS + m] = (d != 0.0f) ? s / d : 0.0f;
  }
}

// E = Y - H[0..k]^T beta
__global__ void scn_resid(float* __restrict__ E, const float* __restrict__ Y,
                          const float* __restrict__ H, const float* __restrict__ beta, int k)
{
  int stride = gridDim.x * blockDim.x;
  for (int i = blockIdx.x * blockDim.x + threadIdx.x; i < NROW; i += stride) {
    float acc[MCLS];
#pragma unroll
    for (int j = 0; j < MCLS; ++j) acc[j] = Y[(size_t)i * MCLS + j];
    for (int t = 0; t <= k; ++t) {
      float h = H[(size_t)t * NROW + i];
#pragma unroll
      for (int j = 0; j < MCLS; ++j) acc[j] -= h * beta[(size_t)t * MCLS + j];
    }
#pragma unroll
    for (int j = 0; j < MCLS; ++j) E[(size_t)i * MCLS + j] = acc[j];
  }
}

// ------------------------------- host side ---------------------------------

extern "C" void kernel_launch(void* const* d_in, const int* in_sizes, int n_in,
                              void* d_out, int out_size, void* d_ws, size_t ws_size,
                              hipStream_t stream)
{
  (void)in_sizes; (void)n_in; (void)out_size; (void)ws_size;
  const float* X = (const float*)d_in[0];   // [10000*784]
  const float* Y = (const float*)d_in[1];   // [10000*10]
  float* Wout    = (float*)d_out;           // [784*64]
  float* bout    = Wout + (size_t)KDIM * MAXN;
  float* betaout = bout + MAXN;             // [64*10]

  unsigned char* base = (unsigned char*)d_ws;
  size_t off = 0;
  auto take = [&](size_t bytes) -> void* {
    void* p = base + off;
    off += (bytes + 255) & ~(size_t)255;
    return p;
  };
  __bf16* Xb   = (__bf16*)take((size_t)NROW * KP * sizeof(__bf16));   // 16.0 MB
  __bf16* Wtb  = (__bf16*)take((size_t)CAND * KP * sizeof(__bf16));   // 256 KB
  float*  bvec = (float*)take(CAND * sizeof(float));
  float*  Hc   = (float*)take((size_t)NROW * CAND * sizeof(float));   // 6.4 MB
  float*  E    = (float*)take((size_t)NROW * MCLS * sizeof(float));
  float*  Hn   = (float*)take((size_t)MAXN * NROW * sizeof(float));   // 2.56 MB
  float*  Qn   = (float*)take((size_t)MAXN * NROW * sizeof(float));   // 2.56 MB
  float*  Rm   = (float*)take((size_t)MAXN * MAXN * sizeof(float));
  float*  QT   = (float*)take((size_t)MAXN * MCLS * sizeof(float));
  float*  beta = (float*)take((size_t)MAXN * MCLS * sizeof(float));
  float*  stats= (float*)take((size_t)(CAND + MCLS * CAND) * sizeof(float));
  float*  ssq  = stats;
  float*  num  = stats + CAND;
  float*  r1s  = (float*)take(72 * sizeof(float));   // r1[64] | scal[8]
  float*  scal = r1s + MAXN;
  int*    dbest= (int*)take(256);

  hipFuncSetAttribute(reinterpret_cast<const void*>(&scn_gemm),
                      hipFuncAttributeMaxDynamicSharedMemorySize, SMEM_BYTES);

  scn_cvt_x<<<(NROW * KP + 255) / 256, 256, 0, stream>>>(X, Xb);
  scn_copy<<<(NROW * MCLS + 255) / 256, 256, 0, stream>>>(E, Y, NROW * MCLS);

  for (int k = 0; k < MAXN; ++k) {
    scn_gen<<<(CAND * KP + 255) / 256, 256, 0, stream>>>(Wtb, bvec, k);
    scn_zero<<<((CAND + MCLS * CAND) + 255) / 256, 256, 0, stream>>>(stats, CAND + MCLS * CAND);
    scn_gemm<<<NBLK, GEMM_THREADS, SMEM_BYTES, stream>>>(Xb, Wtb, bvec, E, Hc, ssq, num);
    scn_select<<<1, 256, 0, stream>>>(num, ssq, Wtb, bvec, Wout, bout, dbest, k);
    scn_gather<<<40, 256, 0, stream>>>(Hn, Hc, dbest, k);
    scn_zero<<<1, 128, 0, stream>>>(r1s, 72);
    if (k > 0) scn_proj<<<40, 256, 0, stream>>>(r1s, Qn, Hn, k);
    scn_maket<<<40, 256, 0, stream>>>(Qn, Hn, r1s, scal, k);
    scn_finr<<<1, 64, 0, stream>>>(Rm, QT, r1s, scal, k);
    scn_normq<<<40, 256, 0, stream>>>(Qn, Y, QT, scal, k);
    scn_solve<<<1, 32, 0, stream>>>(Rm, QT, beta, k);
    scn_resid<<<40, 256, 0, stream>>>(E, Y, Hn, beta, k);
  }
  scn_copy<<<(MAXN * MCLS + 255) / 256, 256, 0, stream>>>(betaout, beta, MAXN * MCLS);
}
